// linoss_pytorch_on_10213432230014
// MI455X (gfx1250) — compile-verified, matrix operand order checked by reading, not run on hardware
//
#include <hip/hip_runtime.h>
#include <math.h>

// ---------------------------------------------------------------------------
// MI455X (gfx1250) LinOSS forward. wave32, WMMA f32_16x16x32_f16 GEMM core
// with double-buffered LDS pipeline + global prefetch.
// ---------------------------------------------------------------------------

typedef _Float16 v16h __attribute__((ext_vector_type(16)));
typedef _Float16 v8h  __attribute__((ext_vector_type(8)));
typedef float    v8f  __attribute__((ext_vector_type(8)));

static __device__ __forceinline__ v8f wmma_f16(v16h a, v16h b, v8f c) {
  // (neg_a, A, neg_b, B, c_mod, C, reuse_a, reuse_b)
  return __builtin_amdgcn_wmma_f32_16x16x32_f16(false, a, false, b, (short)0, c,
                                                false, false);
}

// ------------------------- f32 -> f16 conversion ---------------------------
__global__ void cvt_kernel(const float* __restrict__ in, _Float16* __restrict__ out,
                           float scale, int n) {
  int i = blockIdx.x * blockDim.x + threadIdx.x;
  if (i < n) out[i] = (_Float16)(in[i] * scale);
}

// ------------------------------- LayerNorm ---------------------------------
__global__ __launch_bounds__(256) void layernorm_kernel(
    const float* __restrict__ h, const float* __restrict__ g,
    const float* __restrict__ b, float* __restrict__ hnf,
    _Float16* __restrict__ hn16) {
  __shared__ float r1[256];
  __shared__ float r2[256];
  const int tid = threadIdx.x;
  const size_t base = (size_t)blockIdx.x * 512;
  float x0 = h[base + tid];
  float x1 = h[base + tid + 256];
  r1[tid] = x0 + x1;
  r2[tid] = x0 * x0 + x1 * x1;
  __syncthreads();
  for (int s = 128; s > 0; s >>= 1) {
    if (tid < s) { r1[tid] += r1[tid + s]; r2[tid] += r2[tid + s]; }
    __syncthreads();
  }
  const float mean = r1[0] * (1.0f / 512.0f);
  const float var = r2[0] * (1.0f / 512.0f) - mean * mean;
  const float rstd = rsqrtf(var + 1e-5f);
  float y0 = (x0 - mean) * rstd * g[tid] + b[tid];
  float y1 = (x1 - mean) * rstd * g[tid + 256] + b[tid + 256];
  hnf[base + tid] = y0;
  hnf[base + tid + 256] = y1;
  hn16[base + tid] = (_Float16)y0;
  hn16[base + tid + 256] = (_Float16)y1;
}

// ------------------------------ LinOSS scan --------------------------------
// s_t = M_n * s_{t-1} + f_t with constant 2x2 M per channel. 3-phase chunked
// scan (local scans / chunk composition via M^Lc / correction pass).

static __device__ __forceinline__ void linoss_coefs(
    const float* __restrict__ A_diag, const float* __restrict__ log_steps,
    int n, float& m11, float& m12, float& m21, float& m22, float& dt) {
  float A = fmaxf(A_diag[n], 0.0f);
  dt = 1.0f / (1.0f + expf(-log_steps[n]));
  float S = 1.0f / (1.0f + dt * dt * A);
  m11 = 1.0f - dt * dt * A * S;
  m12 = -dt * A * S;
  m21 = dt * S;
  m22 = S;
}

__global__ __launch_bounds__(512) void scan_phase1(
    const float* __restrict__ A_diag, const float* __restrict__ log_steps,
    float* __restrict__ bu_re, float* __restrict__ bu_im,
    float4* __restrict__ sum_buf, int L, int NCH, int Lc) {
  const int n = threadIdx.x;
  const int b = blockIdx.x / NCH;
  const int ch = blockIdx.x % NCH;
  float m11, m12, m21, m22, dt;
  linoss_coefs(A_diag, log_steps, n, m11, m12, m21, m22, dt);
  const float c1 = m11 * dt, c2 = m21 * dt;
  float zr = 0.f, xr = 0.f, zi = 0.f, xi = 0.f;
  size_t idx = ((size_t)b * L + (size_t)ch * Lc) * 512 + n;
  for (int t = 0; t < Lc; t++) {
    float br_ = bu_re[idx];
    float bi_ = bu_im[idx];
    float nzr = m11 * zr + m12 * xr + c1 * br_;
    float nxr = m21 * zr + m22 * xr + c2 * br_;
    float nzi = m11 * zi + m12 * xi + c1 * bi_;
    float nxi = m21 * zi + m22 * xi + c2 * bi_;
    zr = nzr; xr = nxr; zi = nzi; xi = nxi;
    bu_re[idx] = xr;   // store local x (real)
    bu_im[idx] = xi;   // store local x (imag)
    idx += 512;
  }
  sum_buf[((size_t)b * NCH + ch) * 512 + n] = make_float4(zr, xr, zi, xi);
}

__global__ __launch_bounds__(512) void scan_phase2(
    const float* __restrict__ A_diag, const float* __restrict__ log_steps,
    const float4* __restrict__ sum_buf, float4* __restrict__ start_buf,
    int NCH, int nsq) {
  const int n = threadIdx.x;
  const int b = blockIdx.x;
  float m11, m12, m21, m22, dt;
  linoss_coefs(A_diag, log_steps, n, m11, m12, m21, m22, dt);
  float p11 = m11, p12 = m12, p21 = m21, p22 = m22;
  for (int s = 0; s < nsq; s++) {
    float a = p11 * p11 + p12 * p21;
    float bq = p11 * p12 + p12 * p22;
    float c = p21 * p11 + p22 * p21;
    float d = p21 * p12 + p22 * p22;
    p11 = a; p12 = bq; p21 = c; p22 = d;
  }
  float zr = 0.f, xr = 0.f, zi = 0.f, xi = 0.f;
  for (int c = 0; c < NCH; c++) {
    size_t o = ((size_t)b * NCH + c) * 512 + n;
    start_buf[o] = make_float4(zr, xr, zi, xi);
    float4 e = sum_buf[o];
    float nzr = p11 * zr + p12 * xr + e.x;
    float nxr = p21 * zr + p22 * xr + e.y;
    float nzi = p11 * zi + p12 * xi + e.z;
    float nxi = p21 * zi + p22 * xi + e.w;
    zr = nzr; xr = nxr; zi = nzi; xi = nxi;
  }
}

__global__ __launch_bounds__(512) void scan_phase3(
    const float* __restrict__ A_diag, const float* __restrict__ log_steps,
    const float* __restrict__ lx_re, const float* __restrict__ lx_im,
    const float4* __restrict__ start_buf,
    _Float16* __restrict__ xs_re, _Float16* __restrict__ xs_im,
    int L, int NCH, int Lc) {
  const int n = threadIdx.x;
  const int b = blockIdx.x / NCH;
  const int ch = blockIdx.x % NCH;
  float m11, m12, m21, m22, dt;
  linoss_coefs(A_diag, log_steps, n, m11, m12, m21, m22, dt);
  float4 s = start_buf[((size_t)b * NCH + ch) * 512 + n];
  float p11 = m11, p12 = m12, p21 = m21, p22 = m22;  // P = M^(tau+1)
  size_t idx = ((size_t)b * L + (size_t)ch * Lc) * 512 + n;
  for (int t = 0; t < Lc; t++) {
    float xr = lx_re[idx] + p21 * s.x + p22 * s.y;
    float xi = lx_im[idx] + p21 * s.z + p22 * s.w;
    xs_re[idx] = (_Float16)xr;
    xs_im[idx] = (_Float16)xi;
    float a = m11 * p11 + m12 * p21;
    float bq = m11 * p12 + m12 * p22;
    float c = m21 * p11 + m22 * p21;
    float d = m21 * p12 + m22 * p22;
    p11 = a; p12 = bq; p21 = c; p22 = d;
    idx += 512;
  }
}

// ------------------------------- WMMA GEMM ---------------------------------
// out[m][n] = sum_k A[m][k] * W[n][k]  (+ fused epilogue).
// 128x128 block tile, BK=32, 256 threads = 8 waves (4x2), 2x4 WMMA tiles per
// wave. Double-buffered LDS: iteration k issues global loads for tile k+1
// into staging VGPRs, runs ds_load+WMMA on the current buffer (hiding global
// latency), then ds_stores the staged tile; one barrier per iteration.
template <int EPI>
__global__ __launch_bounds__(256) void gemm_kernel(
    const _Float16* __restrict__ A0g, const _Float16* __restrict__ A1g,
    const _Float16* __restrict__ B0g, const _Float16* __restrict__ B1g,
    const float* __restrict__ bias0, const float* __restrict__ bias1,
    const float* __restrict__ aux0, const float* __restrict__ aux1,
    void* __restrict__ out0v, void* __restrict__ out1v,
    int M, int N, int K) {
  constexpr bool DUAL_A = (EPI == 3);
  constexpr bool DUAL_B = (EPI == 2 || EPI == 3 || EPI == 4);
  constexpr bool DUAL_ACC = (EPI == 2 || EPI == 4);
  constexpr int BM = 128, BN = 128, BK = 32;
  constexpr int TILE = BM * BK;                      // halves per tile
  constexpr int NT = 2 + (DUAL_A ? 1 : 0) + (DUAL_B ? 1 : 0);
  constexpr int oA0 = 0;
  constexpr int oA1 = TILE;                          // valid iff DUAL_A
  constexpr int oB0 = (1 + (DUAL_A ? 1 : 0)) * TILE;
  constexpr int oB1 = oB0 + TILE;                    // valid iff DUAL_B

  __shared__ __align__(16) _Float16 smem[2 * NT * TILE];

  const int tid = threadIdx.x;
  const int lane = tid & 31;
  const int wave = tid >> 5;
  const int waveM = (wave >> 1) * 32;  // 4 wave rows x 32
  const int waveN = (wave & 1) * 64;   // 2 wave cols x 64
  const int blockM = blockIdx.y * BM;
  const int blockN = blockIdx.x * BN;
  const int hi8 = lane >> 4;
  const int l15 = lane & 15;

  v8f acc0[2][4];
  v8f acc1[2][4];
#pragma unroll
  for (int i = 0; i < 2; i++)
#pragma unroll
    for (int j = 0; j < 4; j++) { acc0[i][j] = {}; acc1[i][j] = {}; }

  // Each thread moves two 16B chunks per tile: chunk c -> row c>>2, k (c&3)*8.
  const int c0 = tid, c1 = tid + 256;
  const int r0 = c0 >> 2, kk0 = (c0 & 3) * 8;
  const int r1 = c1 >> 2, kk1 = (c1 & 3) * 8;

  // Issue global loads for a tile into staging regs (no LDS traffic yet).
  // Also prefetch the following k-tile (global_prefetch_b8, probe-verified).
  auto loadG = [&](const _Float16* g, int rowBase, int ldg, int k0, v8h st[2]) {
    const _Float16* p0 = g + (size_t)(rowBase + r0) * ldg + k0 + kk0;
    const _Float16* p1 = g + (size_t)(rowBase + r1) * ldg + k0 + kk1;
    st[0] = *(const v8h*)p0;
    st[1] = *(const v8h*)p1;
    if (k0 + BK < K) {
      __builtin_prefetch(p0 + BK, 0, 1);
      __builtin_prefetch(p1 + BK, 0, 1);
    }
  };
  auto storeL = [&](_Float16* s, const v8h st[2]) {
    *(v8h*)(s + r0 * BK + kk0) = st[0];
    *(v8h*)(s + r1 * BK + kk1) = st[1];
  };
  // Operand fetch: ISA 16-bit 16x32 layout -> lane<16 holds K{0..7,16..23},
  // lane>=16 holds K{8..15,24..31}; two ds_read_b128 per operand.
  auto fetch = [&](const _Float16* s, int rowOff) -> v16h {
    const _Float16* p = s + (rowOff + l15) * BK + hi8 * 8;
    v8h lo = *(const v8h*)(p);
    v8h hc = *(const v8h*)(p + 16);
    return __builtin_shufflevector(lo, hc, 0, 1, 2, 3, 4, 5, 6, 7, 8, 9, 10,
                                   11, 12, 13, 14, 15);
  };

  v8h stA0[2], stA1[2], stB0[2], stB1[2];

  // Prologue: tile 0 -> buffer 0.
  loadG(A0g, blockM, K, 0, stA0);
  if constexpr (DUAL_A) loadG(A1g, blockM, K, 0, stA1);
  loadG(B0g, blockN, K, 0, stB0);
  if constexpr (DUAL_B) loadG(B1g, blockN, K, 0, stB1);
  storeL(smem + oA0, stA0);
  if constexpr (DUAL_A) storeL(smem + oA1, stA1);
  storeL(smem + oB0, stB0);
  if constexpr (DUAL_B) storeL(smem + oB1, stB1);
  __syncthreads();

  const int nk = K / BK;
  for (int kk = 0; kk < nk; kk++) {
    const int cur = kk & 1;
    _Float16* curBuf = smem + cur * (NT * TILE);
    _Float16* nxtBuf = smem + (1 - cur) * (NT * TILE);
    const bool hasNext = (kk + 1 < nk);

    if (hasNext) {  // issue next tile's global loads early
      const int k0 = (kk + 1) * BK;
      loadG(A0g, blockM, K, k0, stA0);
      if constexpr (DUAL_A) loadG(A1g, blockM, K, k0, stA1);
      loadG(B0g, blockN, K, k0, stB0);
      if constexpr (DUAL_B) loadG(B1g, blockN, K, k0, stB1);
    }

    v16h a0[2], a1[2];
#pragma unroll
    for (int i = 0; i < 2; i++) {
      a0[i] = fetch(curBuf + oA0, waveM + i * 16);
      if constexpr (DUAL_A) a1[i] = fetch(curBuf + oA1, waveM + i * 16);
    }
#pragma unroll
    for (int j = 0; j < 4; j++) {
      v16h b0j = fetch(curBuf + oB0, waveN + j * 16);
      v16h b1j{};
      if constexpr (DUAL_B) b1j = fetch(curBuf + oB1, waveN + j * 16);
#pragma unroll
      for (int i = 0; i < 2; i++) {
        acc0[i][j] = wmma_f16(a0[i], b0j, acc0[i][j]);
        if constexpr (DUAL_A) acc0[i][j] = wmma_f16(a1[i], b1j, acc0[i][j]);
        if constexpr (DUAL_ACC) acc1[i][j] = wmma_f16(a0[i], b1j, acc1[i][j]);
      }
    }

    if (hasNext) {  // s_wait_loadcnt happens here, after the WMMA work
      storeL(nxtBuf + oA0, stA0);
      if constexpr (DUAL_A) storeL(nxtBuf + oA1, stA1);
      storeL(nxtBuf + oB0, stB0);
      if constexpr (DUAL_B) storeL(nxtBuf + oB1, stB1);
    }
    __syncthreads();
  }

  // Epilogue. C/D layout: VGPR v, lanes 0-15 -> (M=v, N=lane);
  // lanes 16-31 -> (M=v+8, N=lane-16).
#pragma unroll
  for (int i = 0; i < 2; i++)
#pragma unroll
    for (int j = 0; j < 4; j++)
#pragma unroll
      for (int v = 0; v < 8; v++) {
        const int row = blockM + waveM + i * 16 + hi8 * 8 + v;
        const int col = blockN + waveN + j * 16 + l15;
        const size_t o = (size_t)row * N + col;
        const float a = acc0[i][j][v];
        if constexpr (EPI == 0) {
          ((_Float16*)out0v)[o] = (_Float16)(a + bias0[col]);
        } else if constexpr (EPI == 1) {
          ((float*)out0v)[o] = a + bias0[col];
        } else if constexpr (EPI == 2) {
          ((float*)out0v)[o] = a;
          ((float*)out1v)[o] = acc1[i][j][v];
        } else if constexpr (EPI == 3) {
          float y = a + aux0[col] * aux1[o];
          float u = y + 0.044715f * y * y * y;
          float gl = 0.5f * y * (1.0f + tanhf(0.7978845608028654f * u));
          ((_Float16*)out0v)[o] = (_Float16)gl;
        } else if constexpr (EPI == 4) {
          float u = a + bias0[col];
          float vv = acc1[i][j][v] + bias1[col];
          float r = u * (1.0f / (1.0f + expf(-vv))) + aux1[o];
          ((_Float16*)out0v)[o] = (_Float16)r;
        } else {  // EPI 6
          ((float*)out0v)[o] = a + bias0[col];
        }
      }
}

// ---------------------------------------------------------------------------
extern "C" void kernel_launch(void* const* d_in, const int* in_sizes, int n_in,
                              void* d_out, int out_size, void* d_ws,
                              size_t ws_size, hipStream_t stream) {
  (void)in_sizes; (void)n_in; (void)out_size; (void)ws_size;

  const float* x        = (const float*)d_in[0];
  const float* W_in     = (const float*)d_in[1];
  const float* b_in     = (const float*)d_in[2];
  const float* W_enc    = (const float*)d_in[3];
  const float* b_enc    = (const float*)d_in[4];
  const float* ln_g     = (const float*)d_in[5];
  const float* ln_b     = (const float*)d_in[6];
  const float* A_diag   = (const float*)d_in[7];
  const float* log_steps= (const float*)d_in[8];
  const float* B_re     = (const float*)d_in[9];
  const float* B_im     = (const float*)d_in[10];
  const float* C_re     = (const float*)d_in[11];
  const float* C_im     = (const float*)d_in[12];
  const float* Dvec     = (const float*)d_in[13];
  const float* glu_w1   = (const float*)d_in[14];
  const float* glu_b1   = (const float*)d_in[15];
  const float* glu_w2   = (const float*)d_in[16];
  const float* glu_b2   = (const float*)d_in[17];
  const float* W_dec    = (const float*)d_in[18];
  const float* b_dec    = (const float*)d_in[19];
  const float* W_out    = (const float*)d_in[20];
  const float* b_out    = (const float*)d_in[21];

  enum { Bb = 8, L = 2048, H = 128, Dm = 512 };
  const int Mr = Bb * L;                      // 16384 tokens
  const int NCH = 16, Lc = L / NCH, NSQ = 7;  // Lc = 128 = 2^7

  // ---- workspace carve (bump allocator, 256B aligned) ----
  unsigned char* w = (unsigned char*)d_ws;
  size_t off = 0;
  auto alloc = [&](size_t bytes) -> void* {
    void* p = w + off;
    off += (bytes + 255) & ~(size_t)255;
    return p;
  };
  _Float16* x16    = (_Float16*)alloc((size_t)Mr * H * 2);
  _Float16* Win16  = (_Float16*)alloc((size_t)Dm * H * 2);
  _Float16* Wenc16 = (_Float16*)alloc((size_t)Dm * Dm * 2);
  _Float16* Bre16  = (_Float16*)alloc((size_t)Dm * Dm * 2);
  _Float16* Bim16  = (_Float16*)alloc((size_t)Dm * Dm * 2);
  _Float16* Cre16  = (_Float16*)alloc((size_t)Dm * Dm * 2);
  _Float16* nCim16 = (_Float16*)alloc((size_t)Dm * Dm * 2);  // -C_im
  _Float16* W1_16  = (_Float16*)alloc((size_t)Dm * Dm * 2);
  _Float16* W2_16  = (_Float16*)alloc((size_t)Dm * Dm * 2);
  _Float16* Wdec16 = (_Float16*)alloc((size_t)Dm * Dm * 2);
  _Float16* Wout16 = (_Float16*)alloc((size_t)H * Dm * 2);
  _Float16* h1_16  = (_Float16*)alloc((size_t)Mr * Dm * 2);
  float*    skipf  = (float*)   alloc((size_t)Mr * Dm * 4);  // enc output
  float*    hnf    = (float*)   alloc((size_t)Mr * Dm * 4);
  _Float16* hn16   = (_Float16*)alloc((size_t)Mr * Dm * 2);
  float*    bu_re  = (float*)   alloc((size_t)Mr * Dm * 4);
  float*    bu_im  = (float*)   alloc((size_t)Mr * Dm * 4);
  float4*   sumb   = (float4*)  alloc((size_t)Bb * NCH * Dm * 16);
  float4*   startb = (float4*)  alloc((size_t)Bb * NCH * Dm * 16);
  _Float16* xsre16 = (_Float16*)alloc((size_t)Mr * Dm * 2);
  _Float16* xsim16 = (_Float16*)alloc((size_t)Mr * Dm * 2);
  _Float16* g16    = (_Float16*)alloc((size_t)Mr * Dm * 2);
  _Float16* h2_16  = (_Float16*)alloc((size_t)Mr * Dm * 2);
  _Float16* h3_16  = (_Float16*)alloc((size_t)Mr * Dm * 2);

  auto cvt = [&](const float* src, _Float16* dst, float s, int n) {
    cvt_kernel<<<(n + 255) / 256, 256, 0, stream>>>(src, dst, s, n);
  };
  cvt(x, x16, 1.f, Mr * H);
  cvt(W_in, Win16, 1.f, Dm * H);
  cvt(W_enc, Wenc16, 1.f, Dm * Dm);
  cvt(B_re, Bre16, 1.f, Dm * Dm);
  cvt(B_im, Bim16, 1.f, Dm * Dm);
  cvt(C_re, Cre16, 1.f, Dm * Dm);
  cvt(C_im, nCim16, -1.f, Dm * Dm);  // fold the minus sign of Re(.) here
  cvt(glu_w1, W1_16, 1.f, Dm * Dm);
  cvt(glu_w2, W2_16, 1.f, Dm * Dm);
  cvt(W_dec, Wdec16, 1.f, Dm * Dm);
  cvt(W_out, Wout16, 1.f, H * Dm);

  const dim3 blk(256);
  const dim3 gD(Dm / 128, Mr / 128);   // N=512 GEMMs
  const dim3 gH(H / 128, Mr / 128);    // N=128 GEMM (out-proj)

  // 1) h1 = x @ W_in.T + b_in            (M=16384, N=512, K=128) -> f16
  gemm_kernel<0><<<gD, blk, 0, stream>>>(x16, nullptr, Win16, nullptr, b_in,
                                         nullptr, nullptr, nullptr, h1_16,
                                         nullptr, Mr, Dm, H);
  // 2) skip = h1 @ W_enc.T + b_enc       -> f32
  gemm_kernel<1><<<gD, blk, 0, stream>>>(h1_16, nullptr, Wenc16, nullptr,
                                         b_enc, nullptr, nullptr, nullptr,
                                         skipf, nullptr, Mr, Dm, Dm);
  // 3) layernorm -> hnf (f32) + hn16 (f16)
  layernorm_kernel<<<Mr, 256, 0, stream>>>(skipf, ln_g, ln_b, hnf, hn16);
  // 4) Bu_re = hn @ B_re.T ; Bu_im = hn @ B_im.T  (dual-B dual-acc) -> f32
  gemm_kernel<2><<<gD, blk, 0, stream>>>(hn16, nullptr, Bre16, Bim16, nullptr,
                                         nullptr, nullptr, nullptr, bu_re,
                                         bu_im, Mr, Dm, Dm);
  // 5) chunked LinOSS scan (constant 2x2 per channel)
  scan_phase1<<<Bb * NCH, 512, 0, stream>>>(A_diag, log_steps, bu_re, bu_im,
                                            sumb, L, NCH, Lc);
  scan_phase2<<<Bb, 512, 0, stream>>>(A_diag, log_steps, sumb, startb, NCH,
                                      NSQ);
  scan_phase3<<<Bb * NCH, 512, 0, stream>>>(A_diag, log_steps, bu_re, bu_im,
                                            startb, xsre16, xsim16, L, NCH,
                                            Lc);
  // 6) g = gelu(xs_re@C_re.T - xs_im@C_im.T + D*hn)  (dual-A dual-B) -> f16
  gemm_kernel<3><<<gD, blk, 0, stream>>>(xsre16, xsim16, Cre16, nCim16,
                                         nullptr, nullptr, Dvec, hnf, g16,
                                         nullptr, Mr, Dm, Dm);
  // 7) h2 = skip + (g@w1.T+b1)*sigmoid(g@w2.T+b2)    (dual-B dual-acc) -> f16
  gemm_kernel<4><<<gD, blk, 0, stream>>>(g16, nullptr, W1_16, W2_16, glu_b1,
                                         glu_b2, nullptr, skipf, h2_16,
                                         nullptr, Mr, Dm, Dm);
  // 8) h3 = h2 @ W_dec.T + b_dec -> f16
  gemm_kernel<0><<<gD, blk, 0, stream>>>(h2_16, nullptr, Wdec16, nullptr,
                                         b_dec, nullptr, nullptr, nullptr,
                                         h3_16, nullptr, Mr, Dm, Dm);
  // 9) out = h3 @ W_out.T + b_out        (N=128) -> f32 d_out
  gemm_kernel<6><<<gH, blk, 0, stream>>>(h3_16, nullptr, Wout16, nullptr,
                                         b_out, nullptr, nullptr, nullptr,
                                         d_out, nullptr, Mr, H, Dm);
}
